// GraphAttentionLayer_9861244911818
// MI455X (gfx1250) — compile-verified, matrix operand order checked by reading, not run on hardware
//
#include <hip/hip_runtime.h>
#include <hip/hip_bf16.h>

typedef __attribute__((ext_vector_type(16))) __bf16 v16bf;
typedef __attribute__((ext_vector_type(8)))  __bf16 v8bf;
typedef __attribute__((ext_vector_type(8)))  float  v8f;

#define NN   4096   // nodes
#define INF_ 256    // in features
#define HOUT 256    // H*D
#define NH   4      // heads
#define DD   64     // head dim

union V16 { v16bf v; __bf16 e[16]; v8bf h[2]; };
union V8F { v8f v; float f[8]; };

// ---------------------------------------------------------------------------
// Kernel 1: h = x @ W  (bf16 WMMA, f32 accumulate). One wave per 16x16 tile.
// A layout (16x32 bf16): lane r=l%16 holds row r; K runs {k0..k0+7} and
// {k0+16..k0+23} with k0 = 32*kb + 8*(l/16).
// ---------------------------------------------------------------------------
__global__ __launch_bounds__(128) void gat_gemm1(const float* __restrict__ x,
                                                 const float* __restrict__ W,
                                                 float* __restrict__ h) {
  const int lane = threadIdx.x & 31;
  const int wave = blockIdx.x * 4 + (threadIdx.x >> 5);   // 0..4095
  const int mt = wave >> 4;                               // 0..255
  const int nt = wave & 15;                               // 0..15
  const int i0 = mt * 16, n0 = nt * 16;
  const int r    = lane & 15;
  const int half = lane >> 4;

  V8F c = {};
  for (int kb = 0; kb < 8; ++kb) {
    const int k0 = kb * 32 + half * 8;
    V16 A;                                   // x rows, row-major -> contiguous
    const float* ax = x + (i0 + r) * INF_ + k0;
    #pragma unroll
    for (int kk = 0; kk < 8; ++kk) {
      A.e[kk]     = (__bf16)ax[kk];
      A.e[8 + kk] = (__bf16)ax[16 + kk];
    }
    V16 B;                                   // W column n0+r, strided
    const float* wx = W + k0 * HOUT + (n0 + r);
    #pragma unroll
    for (int kk = 0; kk < 8; ++kk) {
      B.e[kk]     = (__bf16)wx[kk * HOUT];
      B.e[8 + kk] = (__bf16)wx[(16 + kk) * HOUT];
    }
    c.v = __builtin_amdgcn_wmma_f32_16x16x32_bf16(false, A.v, false, B.v,
                                                  (short)0, c.v, false, false);
  }
  #pragma unroll
  for (int v = 0; v < 8; ++v) {              // C layout: row = v + 8*half, col = r
    const int row = v + 8 * half;
    h[(i0 + row) * HOUT + n0 + r] = c.f[v];
  }
}

// ---------------------------------------------------------------------------
// Kernel 2: si/sj scores, stored head-major: siT[hd*NN + i]
// ---------------------------------------------------------------------------
__global__ void gat_scores(const float* __restrict__ h,
                           const float* __restrict__ a1,
                           const float* __restrict__ a2,
                           float* __restrict__ siT, float* __restrict__ sjT) {
  const int t  = blockIdx.x * blockDim.x + threadIdx.x;   // 0..16383
  const int hd = t >> 12;
  const int i  = t & (NN - 1);
  const float* hp = h + i * HOUT + hd * DD;
  float s1 = 0.f, s2 = 0.f;
  for (int d = 0; d < DD; ++d) { float v = hp[d]; s1 += v * a1[d]; s2 += v * a2[d]; }
  siT[t] = s1; sjT[t] = s2;
}

// ---------------------------------------------------------------------------
// Kernel 3: pack h into WMMA B-matrix layout (bf16), tile = (head, jb, dt):
// 32 (j) x 16 (d). Lane's 16 bf16 stored contiguously so the attention
// kernel loads each B operand with two global_load_b128.
// ---------------------------------------------------------------------------
__global__ __launch_bounds__(128) void gat_pack(const float* __restrict__ h,
                                                __bf16* __restrict__ hB) {
  const int lane = threadIdx.x & 31;
  const int wave = blockIdx.x * 4 + (threadIdx.x >> 5);   // 0..2047
  const int dt = wave & 3;
  const int jb = (wave >> 2) & 127;
  const int hd = wave >> 9;
  const int n     = lane & 15;
  const int kbase = (lane >> 4) * 8;
  V16 B;
  #pragma unroll
  for (int kk = 0; kk < 8; ++kk) {
    B.e[kk]     = (__bf16)h[(jb * 32 + kbase + kk) * HOUT + hd * DD + dt * 16 + n];
    B.e[8 + kk] = (__bf16)h[(jb * 32 + kbase + 16 + kk) * HOUT + hd * DD + dt * 16 + n];
  }
  __bf16* dst = hB + (size_t)((hd * 128 + jb) * 4 + dt) * 512 + lane * 16;
  *(v8bf*)dst       = B.h[0];
  *(v8bf*)(dst + 8) = B.h[1];
}

// ---------------------------------------------------------------------------
// Kernel 4: per-head max of sj (softmax stability bound). 1 wave per head.
// ---------------------------------------------------------------------------
__global__ __launch_bounds__(128) void gat_sjmax(const float* __restrict__ sjT,
                                                 float* __restrict__ sjmax) {
  const int hd = threadIdx.x >> 5;
  const int lane = threadIdx.x & 31;
  float m = -1e30f;
  for (int i = lane; i < NN; i += 32) m = fmaxf(m, sjT[hd * NN + i]);
  #pragma unroll
  for (int off = 16; off >= 1; off >>= 1) m = fmaxf(m, __shfl_xor(m, off, 32));
  if (lane == 0) sjmax[hd] = m;
}

// ---------------------------------------------------------------------------
// Kernel 5: fused mask -> exp -> aggregate. 4 waves per (16-row i-tile, head);
// wave w covers j-blocks [w*32, w*32+32). Fixed bound-max
// m_i = lrelu(si_i + max_j sj_j) >= every row score, so exp never overflows,
// no online rescaling, and partials across waves combine by pure addition
// (LDS reduction at the end). 4 WMMAs per 32-wide j-block per wave.
// ---------------------------------------------------------------------------
__global__ __launch_bounds__(128) void gat_attn(const int* __restrict__ adj,
                                                const float* __restrict__ siT,
                                                const float* __restrict__ sjT,
                                                const float* __restrict__ sjmax,
                                                const __bf16* __restrict__ hB,
                                                float* __restrict__ out) {
  const int tid   = threadIdx.x;
  const int lane  = tid & 31;
  const int w     = tid >> 5;                    // wave 0..3
  const int itile = blockIdx.x;                  // 0..255
  const int hd    = blockIdx.y;                  // 0..3
  const int i0    = itile * 16;
  const int r     = lane & 15;
  const int half  = lane >> 4;

  __shared__ float sjL[NN];                      // 16 KB: sj for this head
  __shared__ float cL[4][16][68];                // ~17 KB partial C (padded)
  __shared__ float lsumL[4][16];
  {
    const float4* src = (const float4*)(sjT + hd * NN);
    float4* dst = (float4*)sjL;
    for (int q = tid; q < NN / 4; q += 128) dst[q] = src[q];
  }
  __syncthreads();

  const float si_r = siT[hd * NN + i0 + r];
  float m_r = si_r + sjmax[hd];
  m_r = m_r > 0.f ? m_r : 0.2f * m_r;            // lrelu of the upper bound

  float lsum = 0.f;
  V8F c0 = {}, c1 = {}, c2 = {}, c3 = {};

  const int*    adjRow = adj + (size_t)(i0 + r) * NN;
  const __bf16* hBh    = hB + (size_t)hd * 128 * 4 * 512 + lane * 16;

  const int jb0 = w * 32, jb1 = jb0 + 32;        // this wave's j-range
  for (int jb = jb0; jb < jb1; ++jb) {
    const int jc = jb * 32 + half * 8;           // first K run start (A layout)

    // pull next adj cachelines toward the WGP while we do exp/WMMA work
    __builtin_prefetch(adjRow + jc + 256, 0, 3);

    int4 a0  = *(const int4*)(adjRow + jc);
    int4 a1v = *(const int4*)(adjRow + jc + 4);
    int4 a2v = *(const int4*)(adjRow + jc + 16);
    int4 a3v = *(const int4*)(adjRow + jc + 20);

    float4 s0 = *(const float4*)(sjL + jc);
    float4 s1 = *(const float4*)(sjL + jc + 4);
    float4 s2 = *(const float4*)(sjL + jc + 16);
    float4 s3 = *(const float4*)(sjL + jc + 20);

    const float sv[16] = {s0.x,s0.y,s0.z,s0.w, s1.x,s1.y,s1.z,s1.w,
                          s2.x,s2.y,s2.z,s2.w, s3.x,s3.y,s3.z,s3.w};
    const int   av[16] = {a0.x,a0.y,a0.z,a0.w, a1v.x,a1v.y,a1v.z,a1v.w,
                          a2v.x,a2v.y,a2v.z,a2v.w, a3v.x,a3v.y,a3v.z,a3v.w};

    V16 A;
    #pragma unroll
    for (int kk = 0; kk < 16; ++kk) {
      float e = si_r + sv[kk];
      e = e > 0.f ? e : 0.2f * e;                // leaky_relu
      float p = (av[kk] > 0) ? __expf(e - m_r) : 0.f;
      lsum += p;
      A.e[kk] = (__bf16)p;                       // already in A-matrix layout
    }

    const __bf16* bt = hBh + (size_t)jb * 4 * 512;
    V16 B0, B1, B2, B3;
    B0.h[0] = *(const v8bf*)(bt);        B0.h[1] = *(const v8bf*)(bt + 8);
    B1.h[0] = *(const v8bf*)(bt + 512);  B1.h[1] = *(const v8bf*)(bt + 520);
    B2.h[0] = *(const v8bf*)(bt + 1024); B2.h[1] = *(const v8bf*)(bt + 1032);
    B3.h[0] = *(const v8bf*)(bt + 1536); B3.h[1] = *(const v8bf*)(bt + 1544);

    c0.v = __builtin_amdgcn_wmma_f32_16x16x32_bf16(false, A.v, false, B0.v, (short)0, c0.v, false, false);
    c1.v = __builtin_amdgcn_wmma_f32_16x16x32_bf16(false, A.v, false, B1.v, (short)0, c1.v, false, false);
    c2.v = __builtin_amdgcn_wmma_f32_16x16x32_bf16(false, A.v, false, B2.v, (short)0, c2.v, false, false);
    c3.v = __builtin_amdgcn_wmma_f32_16x16x32_bf16(false, A.v, false, B3.v, (short)0, c3.v, false, false);
  }

  // ---- cross-wave reduction (pure addition thanks to fixed max) ----
  lsum += __shfl_xor(lsum, 16, 32);              // join the two K-run halves
  if (half == 0) lsumL[w][r] = lsum;

  #pragma unroll
  for (int v = 0; v < 8; ++v) {
    const int row = v + 8 * half;                // C layout row
    cL[w][row][ 0 + r] = c0.f[v];
    cL[w][row][16 + r] = c1.f[v];
    cL[w][row][32 + r] = c2.f[v];
    cL[w][row][48 + r] = c3.f[v];
  }
  __syncthreads();

  // wave w finalizes d-tile dt = w
  const int dt = w;
  #pragma unroll
  for (int v = 0; v < 8; ++v) {
    const int row = v + 8 * half;
    const float s  = cL[0][row][dt * 16 + r] + cL[1][row][dt * 16 + r] +
                     cL[2][row][dt * 16 + r] + cL[3][row][dt * 16 + r];
    const float lt = lsumL[0][row] + lsumL[1][row] +
                     lsumL[2][row] + lsumL[3][row];
    out[(size_t)(i0 + row) * HOUT + hd * DD + dt * 16 + r] = s / (lt + 1e-30f);
  }
}

// ---------------------------------------------------------------------------
extern "C" void kernel_launch(void* const* d_in, const int* in_sizes, int n_in,
                              void* d_out, int out_size, void* d_ws, size_t ws_size,
                              hipStream_t stream) {
  const float* x   = (const float*)d_in[0];
  const int*   adj = (const int*)d_in[1];
  const float* W   = (const float*)d_in[2];
  const float* a1  = (const float*)d_in[3];
  const float* a2  = (const float*)d_in[4];
  float* out = (float*)d_out;

  char* ws = (char*)d_ws;
  float*  h    = (float*)ws;                       // 4 MB   f32 h
  __bf16* hB   = (__bf16*)(ws + (4u << 20));       // 2 MB   packed B-layout
  float*  siT  = (float*)(ws + (6u << 20));        // 64 KB
  float*  sjT  = siT + NN * NH;                    // 64 KB
  float*  sjmx = sjT + NN * NH;                    // 16 B

  gat_gemm1 <<<1024, 128, 0, stream>>>(x, W, h);
  gat_scores<<<64, 256, 0, stream>>>(h, a1, a2, siT, sjT);
  gat_pack  <<<512, 128, 0, stream>>>(h, hB);
  gat_sjmax <<<1, 128, 0, stream>>>(sjT, sjmx);
  gat_attn  <<<dim3(256, 4), 128, 0, stream>>>(adj, siT, sjT, sjmx, hB, out);
}
